// baseline_model_38500086841783
// MI455X (gfx1250) — matrix operand order repaired and checked by reading, not compiled, not run on hardware
//
#include <hip/hip_runtime.h>
#include <hip/hip_bf16.h>

// ---------------- problem constants (from reference) ----------------
#define VOCAB 32000
#define EMB   128
#define HID   256
#define G4    1024      // 4*HID
#define BS    8
#define SEQL  1024
#define ROWS  (BS * SEQL)   // 8192

// ---------------- vector types ----------------
typedef __attribute__((ext_vector_type(16))) __bf16 v16bf;
typedef __attribute__((ext_vector_type(8)))  __bf16 v8bf;
typedef __attribute__((ext_vector_type(8)))  float  v8f;
typedef unsigned int u32;
typedef __attribute__((ext_vector_type(4)))  u32 v4u;
typedef __attribute__((ext_vector_type(8)))  u32 v8u;

__device__ __forceinline__ __bf16 f2bf(float x) {
    unsigned int u = __builtin_bit_cast(unsigned int, x);
    unsigned int r = (u + 0x7FFFu + ((u >> 16) & 1u)) >> 16;
    return __builtin_bit_cast(__bf16, (unsigned short)r);
}

__device__ __forceinline__ v16bf join8(v8bf lo, v8bf hi) {
    return __builtin_shufflevector(lo, hi, 0,1,2,3,4,5,6,7,8,9,10,11,12,13,14,15);
}

__device__ __forceinline__ float sigf(float x) {
    return 1.0f / (1.0f + __expf(-x));
}

// ---------------- kernel 0: weight conversion / transposition ----------------
// Wt  [G4][EMB]    bf16 (n-major copy of W  [EMB][G4])
// Ut  [G4][HID]    bf16 (n-major copy of U  [HID][G4])
// wlT [VOCAB][HID] bf16 (v-major copy of w_linear [HID][VOCAB])
__global__ __launch_bounds__(256) void convert_weights(
    const float* __restrict__ W, const float* __restrict__ U,
    const float* __restrict__ wl,
    __bf16* __restrict__ Wt, __bf16* __restrict__ Ut, __bf16* __restrict__ wlT)
{
    long i = (long)blockIdx.x * blockDim.x + threadIdx.x;
    if (i < (long)G4 * EMB) {               // 131072
        int n = (int)(i >> 7), k = (int)(i & 127);
        Wt[i] = f2bf(W[(long)k * G4 + n]);
    }
    if (i < (long)G4 * HID) {               // 262144
        int n = (int)(i >> 8), k = (int)(i & 255);
        Ut[i] = f2bf(U[(long)k * G4 + n]);
    }
    if (i < (long)VOCAB * HID) {            // 8192000
        int v = (int)(i >> 8), h = (int)(i & 255);
        wlT[i] = f2bf(wl[(long)h * VOCAB + v]);
    }
}

// ---------------- kernel 0b: embedding gather -> bf16 activations ----------------
// xbf[r][k] = bf16(emb[sent[r]][k]), coalesced; removes converts from GEMM loop.
__global__ __launch_bounds__(256) void embed_gather_bf(
    const int* __restrict__ sent, const float* __restrict__ emb,
    __bf16* __restrict__ xbf)
{
    long i = (long)blockIdx.x * blockDim.x + threadIdx.x;   // ROWS*EMB
    int r = (int)(i >> 7), k = (int)(i & 127);
    xbf[i] = f2bf(emb[(long)sent[r] * EMB + k]);
}

// ---------------- kernel 1: xW = x@W + bias (bf16 WMMA, f32 accum) ----------------
__global__ __launch_bounds__(256) void embed_gemm_xw(
    const __bf16* __restrict__ xbf,   // ROWS x EMB bf16
    const __bf16* __restrict__ Wt,    // G4 x EMB bf16 (n-major)
    const float* __restrict__ bias,   // G4
    float* __restrict__ xW)           // ROWS x G4 f32
{
    int wave = threadIdx.x >> 5;
    int lane = threadIdx.x & 31;
    long tile = (long)blockIdx.x * 8 + wave;   // 32768 tiles total
    int mtile = (int)(tile >> 6);              // ROWS/16 = 512
    int ntile = (int)(tile & 63);              // G4/16  = 64

    int klo = (lane < 16) ? 0 : 8;
    int kbB = (lane < 16) ? 0 : 16;
    int nl  = lane & 15;
    const __bf16* arow = xbf + (long)(mtile * 16 + nl) * EMB;
    const __bf16* brow = Wt  + (long)(ntile * 16 + nl) * EMB;

    v8f acc = {};
#pragma unroll
    for (int ks = 0; ks < EMB; ks += 32) {
        v8bf lo = *(const v8bf*)(arow + ks + kbB);
        v8bf hi = *(const v8bf*)(arow + ks + 8 + kbB);
        v16bf a = join8(lo, hi);
        v16bf b = *(const v16bf*)(brow + ks + kbB);   // 32B aligned, contiguous K
        acc = __builtin_amdgcn_wmma_f32_16x16x32_bf16(false, a, false, b,
                                                      (short)0, acc, false, false);
    }

    int mo  = (lane < 16) ? 0 : 8;
    int col = ntile * 16 + nl;
    float bv = bias[col];
#pragma unroll
    for (int r = 0; r < 8; ++r) {
        int row = mtile * 16 + mo + r;
        xW[(long)row * G4 + col] = acc[r] + bv;
    }
}

// ---------------- kernel 2: sequential LSTM recurrence ----------------
// Single workgroup, 1024 threads = 32 waves. h,c,gates live in LDS.
__global__ __launch_bounds__(1024) void lstm_recurrence(
    const float* __restrict__ xW,     // ROWS x G4 f32 (row = b*SEQL + s)
    const __bf16* __restrict__ Ut,    // G4 x HID bf16 (n-major)
    __bf16* __restrict__ hseq)        // ROWS x HID bf16 hidden states
{
    __shared__ float  gates[BS * G4];   // 32 KB
    __shared__ __bf16 hbf[16 * HID];    // 8 KB  (rows 8..15 = zero M-padding)
    __shared__ float  cst[BS * HID];    // 8 KB

    int tid  = threadIdx.x;
    int wave = tid >> 5;
    int lane = tid & 31;

    for (int i = tid; i < 16 * HID; i += 1024) hbf[i] = (__bf16)0.0f;
    for (int i = tid; i < BS * HID; i += 1024) cst[i] = 0.0f;
    __syncthreads();

    int klo = (lane < 16) ? 0 : 8;
    int kbB = (lane < 16) ? 0 : 16;
    int nl  = lane & 15;
    int mo  = (lane < 16) ? 0 : 8;

    for (int s = 0; s < SEQL; ++s) {
        // ---- phase 1: gate tiles via WMMA (each wave: 2 N-tiles of 16) ----
#pragma unroll
        for (int t = 0; t < 2; ++t) {
            int nt = wave * 2 + t;                     // 0..63
            const __bf16* brow = Ut + (long)(nt * 16 + nl) * HID;
            v8f acc = {};
#pragma unroll
            for (int ks = 0; ks < HID; ks += 32) {
                v8bf lo = *(const v8bf*)(hbf + nl * HID + ks + kbB);
                v8bf hi = *(const v8bf*)(hbf + nl * HID + ks + 8 + kbB);
                v16bf a = join8(lo, hi);
                v16bf b = *(const v16bf*)(brow + ks + kbB);
                acc = __builtin_amdgcn_wmma_f32_16x16x32_bf16(false, a, false, b,
                                                              (short)0, acc, false, false);
            }
            int col = nt * 16 + nl;
            if (mo == 0) {                              // rows 0..7 = valid batches
#pragma unroll
                for (int r = 0; r < 8; ++r) {
                    gates[r * G4 + col] =
                        acc[r] + xW[((long)r * SEQL + s) * G4 + col];
                }
            }
        }
        __syncthreads();

        // ---- phase 2: pointwise LSTM cell update (2048 (b,u) pairs) ----
#pragma unroll
        for (int it = 0; it < 2; ++it) {
            int j = it * 1024 + tid;                    // 0..2047
            int b = j >> 8, u = j & 255;
            float gi = gates[b * G4 + u];
            float gf = gates[b * G4 + HID + u];
            float gg = gates[b * G4 + 2 * HID + u];
            float go = gates[b * G4 + 3 * HID + u];
            float iv = sigf(gi), fv = sigf(gf), gv = tanhf(gg), ov = sigf(go);
            float c = fv * cst[b * HID + u] + iv * gv;
            cst[b * HID + u] = c;
            float h = ov * tanhf(c);
            __bf16 hb = f2bf(h);
            hbf[b * HID + u] = hb;
            hseq[((long)b * SEQL + s) * HID + u] = hb;
        }
        __syncthreads();
    }
}

// ---------------- kernel 3: output projection (dominant GEMM) ----------------
// block = 256 thr (8 waves) computes 128(M) x 64(N).
// B tile (64x256 bf16 = 32KB) is DMA'd into LDS by the Tensor Data Mover.
__global__ __launch_bounds__(256) void out_gemm(
    const __bf16* __restrict__ hseq,  // ROWS x HID bf16
    const __bf16* __restrict__ wlT,   // VOCAB x HID bf16 (v-major)
    const float* __restrict__ biasl,  // VOCAB
    float* __restrict__ out)          // ROWS x VOCAB f32
{
    __shared__ __bf16 Bs[64 * HID];   // 32 KB

    int tid = threadIdx.x;
    int mgroup = blockIdx.x;          // 0..63   (ROWS/128)
    int ngroup = blockIdx.y;          // 0..499  (VOCAB/64)

    // ---- TDM: 2D tile (tile_dim0=256 elems, tile_dim1=64 rows, 2B/elem)
    //      from wlT[ngroup*64 .. +64)[0..256) into LDS Bs. Wave 0 issues;
    //      TENSORcnt tracks completion; barrier publishes to the block.
    if (tid < 32) {
        unsigned long long gaddr =
            (unsigned long long)(const void*)(wlT + (long)ngroup * 64 * HID);
        u32 lds_addr = (u32)(unsigned long long)(void*)Bs;  // flat[31:0] == LDS offset

        v4u g0;
        g0[0] = 1u;                                        // count=1 (valid), user mode
        g0[1] = lds_addr;                                  // lds_addr [63:32]
        g0[2] = (u32)gaddr;                                // global_addr [95:64]
        g0[3] = ((u32)(gaddr >> 32) & 0x01FFFFFFu)         // global_addr [120:96]
                | (2u << 30);                              // type=2 (image) [127:126]
        v8u g1;
        g1[0] = 1u << 16;        // workgroup_mask=0, data_size=1 (2 bytes)
        g1[1] = (u32)HID << 16;  // tensor_dim0[15:0]=256 at bits [63:48]
        g1[2] = 64u << 16;       // tensor_dim0 hi=0, tensor_dim1[15:0]=64
        g1[3] = (u32)HID << 16;  // tensor_dim1 hi=0, tile_dim0=256
        g1[4] = 64u;             // tile_dim1=64, tile_dim2=0
        g1[5] = (u32)HID;        // tensor_dim0_stride[31:0]=256
        g1[6] = 0u;              // stride0 hi=0, tensor_dim1_stride lo=0 (2D)
        g1[7] = 0u;              // tensor_dim1_stride hi=0

        asm volatile("tensor_load_to_lds %0, %1" :: "s"(g0), "s"(g1) : "memory");
        __builtin_amdgcn_s_wait_tensorcnt(0);
    }
    __syncthreads();

    int wave = tid >> 5, lane = tid & 31;
    int klo = (lane < 16) ? 0 : 8;
    int kbB = (lane < 16) ? 0 : 16;
    int nl  = lane & 15;
    int mrow = mgroup * 128 + wave * 16 + nl;
    const __bf16* arow = hseq + (long)mrow * HID;
    __builtin_prefetch(arow, 0, 0);   // global_prefetch_b8 on the streamed A row

    v8f acc[4] = {};
#pragma unroll
    for (int ks = 0; ks < HID; ks += 32) {
        v8bf lo = *(const v8bf*)(arow + ks + kbB);
        v8bf hi = *(const v8bf*)(arow + ks + 8 + kbB);
        v16bf a = join8(lo, hi);
#pragma unroll
        for (int nt = 0; nt < 4; ++nt) {
            v16bf b = *(const v16bf*)(Bs + (nt * 16 + nl) * HID + ks + kbB);
            acc[nt] = __builtin_amdgcn_wmma_f32_16x16x32_bf16(false, a, false, b,
                                                              (short)0, acc[nt], false, false);
        }
    }

    int mo = (lane < 16) ? 0 : 8;
#pragma unroll
    for (int nt = 0; nt < 4; ++nt) {
        int col = ngroup * 64 + nt * 16 + nl;
        float bv = biasl[col];
#pragma unroll
        for (int r = 0; r < 8; ++r) {
            int row = mgroup * 128 + wave * 16 + mo + r;
            out[(long)row * VOCAB + col] = acc[nt][r] + bv;
        }
    }
}

// ---------------- launcher ----------------
extern "C" void kernel_launch(void* const* d_in, const int* in_sizes, int n_in,
                              void* d_out, int out_size, void* d_ws, size_t ws_size,
                              hipStream_t stream) {
    const int*   sent  = (const int*)  d_in[0];
    const float* emb   = (const float*)d_in[1];
    const float* W     = (const float*)d_in[2];
    const float* U     = (const float*)d_in[3];
    const float* bias  = (const float*)d_in[4];
    const float* wl    = (const float*)d_in[5];
    const float* biasl = (const float*)d_in[6];
    float* out = (float*)d_out;

    char* p = (char*)d_ws;
    float*  xW   = (float*)p;   p += (size_t)ROWS * G4 * sizeof(float);    // 32 MB
    __bf16* Wt   = (__bf16*)p;  p += (size_t)G4 * EMB * sizeof(__bf16);    // 256 KB
    __bf16* Ut   = (__bf16*)p;  p += (size_t)G4 * HID * sizeof(__bf16);    // 512 KB
    __bf16* wlT  = (__bf16*)p;  p += (size_t)VOCAB * HID * sizeof(__bf16); // 15.6 MB
    __bf16* hseq = (__bf16*)p;  p += (size_t)ROWS * HID * sizeof(__bf16);  // 4 MB
    __bf16* xbf  = (__bf16*)p;  p += (size_t)ROWS * EMB * sizeof(__bf16);  // 2 MB

    // 0) convert + transpose weights to bf16 n-major
    convert_weights<<<VOCAB * HID / 256, 256, 0, stream>>>(W, U, wl, Wt, Ut, wlT);

    // 0b) embedding gather -> bf16 activations (coalesced, one-shot)
    embed_gather_bf<<<ROWS * EMB / 256, 256, 0, stream>>>(sent, emb, xbf);

    // 1) input GEMM: 32768 wave-tiles / 8 waves per block
    embed_gemm_xw<<<(ROWS / 16) * (G4 / 16) / 8, 256, 0, stream>>>(xbf, Wt, bias, xW);

    // 2) sequential recurrence: one WGP-resident workgroup
    lstm_recurrence<<<1, 1024, 0, stream>>>(xW, Ut, hseq);

    // 3) output projection: 134 GFLOP, 1.05 GB logits; TDM-staged B tiles
    dim3 grid(ROWS / 128, VOCAB / 64);
    out_gemm<<<grid, 256, 0, stream>>>(hseq, wlT, biasl, out);
}
